// GRUPredictor_7756710937029
// MI455X (gfx1250) — compile-verified, matrix operand order checked by reading, not run on hardware
//
#include <hip/hip_runtime.h>
#include <math.h>

// GRU recurrence on gfx1250 via V_WMMA_F32_16X16X32_F16.
// B=4096, T=512, IN=11, H=16. One wave (32 threads) owns a 16-row batch tile.
//
// Fragment layout assumptions (CDNA5 ISA 7.12.2, wave32):
//  A (16-bit, 16x32):  lane l: M = l&15; elements j=0..7 hold K = (l<16?0:8)+j;
//                      elements 8..15 are K=16..31 (kept zero).
//  B (16-bit, 32x16):  mirrored: lane l: N = l&15; element j -> K = (l<16?0:8)+j.
//  C/D (f32 16x16):    lane l: N = l&15; element r -> M = r + (l<16?0:8).

typedef __attribute__((ext_vector_type(16))) _Float16 v16h;
typedef __attribute__((ext_vector_type(8)))  float    v8f;

#define B_TOTAL    4096
#define T_STEPS    512
#define IN_DIM     11
#define H_DIM      16
#define LDS_STRIDE 20   // floats; 80B row stride -> 16B-aligned b128 stores, conflict-free reads

// Fast sigmoid/tanh: v_exp_f32 + single v_rcp_f32 (no IEEE div expansion).
__device__ __forceinline__ float fast_sigmoid(float x) {
    return __builtin_amdgcn_rcpf(1.f + __expf(-x));
}
__device__ __forceinline__ float fast_tanh(float x) {
    const float e = __expf(-2.f * fabsf(x));
    return copysignf((1.f - e) * __builtin_amdgcn_rcpf(1.f + e), x);
}

__global__ __launch_bounds__(32)
void gru_wmma_kernel(const float* __restrict__ xin,
                     const float* __restrict__ W_ih,
                     const float* __restrict__ W_hh,
                     const float* __restrict__ b_ih,
                     const float* __restrict__ b_hh,
                     const float* __restrict__ W_out,
                     const float* __restrict__ b_out,
                     float* __restrict__ out)
{
    __shared__ __align__(16) float tile[H_DIM * LDS_STRIDE];  // tile[hidden][batch_row]

    const int  lane  = threadIdx.x & 31;
    const int  m     = lane & 15;        // batch-row (A/D) or column (B/D) index
    const int  khalf = lane >> 4;        // which K-half this lane holds
    const int  k0    = khalf * 8;
    const int  b0    = blockIdx.x * 16;  // first batch row of this tile
    const bool hi    = (khalf != 0);

    // ---------------- loop-invariant weight fragments ----------------
    v16h Bih[3], Bhh[3];
#pragma unroll
    for (int g = 0; g < 3; ++g) {
        const float* hrow = W_hh + (g * 16 + m) * H_DIM + k0;  // B[k][n] = W_hh[n][k]
        const float* irow = W_ih + (g * 16 + m) * IN_DIM;
        v16h bh = {}; v16h bi = {};
#pragma unroll
        for (int j = 0; j < 8; ++j) {
            bh[j] = (_Float16)hrow[j];
            const int k = k0 + j;
            bi[j] = (k < IN_DIM) ? (_Float16)irow[k] : (_Float16)0.f;
        }
        Bhh[g] = bh; Bih[g] = bi;
    }

    // Bias fragments (broadcast along M: identical in all 8 D-elements per lane).
    const float brz_r = b_ih[m]      + b_hh[m];
    const float brz_z = b_ih[16 + m] + b_hh[16 + m];
    const float bxn   = b_ih[32 + m];
    const float bhn   = b_hh[32 + m];
    v8f Cr0 = {}, Cz0 = {}, Cxn0 = {}, Chn0 = {};
#pragma unroll
    for (int r = 0; r < 8; ++r) { Cr0[r] = brz_r; Cz0[r] = brz_z; Cxn0[r] = bxn; Chn0[r] = bhn; }

    float wo[8];
#pragma unroll
    for (int j = 0; j < 8; ++j) wo[j] = W_out[k0 + j];
    const float bo = b_out[0];

    // Per-lane row base for x. Branch-free loader: all loads unconditional & in-bounds.
    //  j=0..2: k = k0+j  (k0+2 <= 10, valid for both halves)
    //  j=3..7: load k=3..7 of the row (valid), select-zero for the high half (v_cndmask)
    const float* xrow = xin + (size_t)(b0 + m) * T_STEPS * IN_DIM;
    auto load_x = [&](int tt, float* xv) {
        const float* p0 = xrow + (size_t)tt * IN_DIM + k0;
        const float* p1 = xrow + (size_t)tt * IN_DIM + 3;
        xv[0] = p0[0]; xv[1] = p0[1]; xv[2] = p0[2];
#pragma unroll
        for (int j = 3; j < 8; ++j) { const float v = p1[j - 3]; xv[j] = hi ? 0.f : v; }
    };

    float xv[8];
    load_x(0, xv);

    v8f  h_prev = {};   // f32 state, D layout
    v16h Ah     = {};   // h as A fragment (f16); upper 8 elements stay zero for all time
    v16h Ax     = {};   // x as A fragment (f16); upper 8 elements stay zero for all time
    float* outp = out + (size_t)(b0 + m) * T_STEPS;

#pragma unroll 1
    for (int t = 0; t < T_STEPS; ++t) {
        // Build A_x from preloaded values (only low 8 elements change).
#pragma unroll
        for (int j = 0; j < 8; ++j) Ax[j] = (_Float16)xv[j];

        // Branch-free software pipeline: clamped time indices keep all addresses in-bounds.
        const int tn = (t + 1 < T_STEPS) ? t + 1 : T_STEPS - 1;  // s_min, no branch
        const int tp = (t + 9 < T_STEPS) ? t + 9 : T_STEPS - 1;
        __builtin_prefetch(xrow + (size_t)tp * IN_DIM, 0, 1);    // global_prefetch_b8
        load_x(tn, xv);

        // Gate pre-activations: D = A*B + C chains (f16 in, f32 accumulate).
        v8f pr = __builtin_amdgcn_wmma_f32_16x16x32_f16(false, Ax, false, Bih[0], (short)0, Cr0,  false, false);
        pr     = __builtin_amdgcn_wmma_f32_16x16x32_f16(false, Ah, false, Bhh[0], (short)0, pr,   false, false);
        v8f pz = __builtin_amdgcn_wmma_f32_16x16x32_f16(false, Ax, false, Bih[1], (short)0, Cz0,  false, false);
        pz     = __builtin_amdgcn_wmma_f32_16x16x32_f16(false, Ah, false, Bhh[1], (short)0, pz,   false, false);
        v8f xn = __builtin_amdgcn_wmma_f32_16x16x32_f16(false, Ax, false, Bih[2], (short)0, Cxn0, false, false);
        v8f hn = __builtin_amdgcn_wmma_f32_16x16x32_f16(false, Ah, false, Bhh[2], (short)0, Chn0, false, false);

        // Elementwise GRU update (f32, D layout): exp + fast rcp only.
        v8f h_new;
#pragma unroll
        for (int r = 0; r < 8; ++r) {
            const float rg = fast_sigmoid(pr[r]);
            const float zg = fast_sigmoid(pz[r]);
            const float ng = fast_tanh(xn[r] + rg * hn[r]);
            h_new[r] = ng + zg * (h_prev[r] - ng);
        }
        h_prev = h_new;

        // D-layout -> LDS (hidden-major): lane writes 8 consecutive floats (2x ds_store_b128).
        {
            float* dst = &tile[m * LDS_STRIDE + khalf * 8];
            reinterpret_cast<float4*>(dst)[0] = make_float4(h_new[0], h_new[1], h_new[2], h_new[3]);
            reinterpret_cast<float4*>(dst)[1] = make_float4(h_new[4], h_new[5], h_new[6], h_new[7]);
        }
        __syncthreads();   // cross-lane RAW: store -> load

        // Read back row-major: lane l needs h[row=m][k0+j].
        float hv[8];
#pragma unroll
        for (int j = 0; j < 8; ++j) hv[j] = tile[(k0 + j) * LDS_STRIDE + m];
        // WAR edge (loads -> next iteration's stores): same-wave DS ops execute in order
        // (ISA 7.1 "LDS: in order"), so a compiler-only scheduling barrier suffices.
        __builtin_amdgcn_wave_barrier();

        // Rebuild A_h for the next step (only low 8 elements change).
#pragma unroll
        for (int j = 0; j < 8; ++j) Ah[j] = (_Float16)hv[j];

        // Output head: out[b0+m, t] = h_row . W_out + b_out (two K-halves via lane-xor-16).
        float p = 0.f;
#pragma unroll
        for (int j = 0; j < 8; ++j) p += hv[j] * wo[j];
        p += __shfl_xor(p, 16, 32);
        if (!hi) outp[t] = p + bo;
    }
}

extern "C" void kernel_launch(void* const* d_in, const int* in_sizes, int n_in,
                              void* d_out, int out_size, void* d_ws, size_t ws_size,
                              hipStream_t stream)
{
    const float* xin   = (const float*)d_in[0];
    const float* W_ih  = (const float*)d_in[1];
    const float* W_hh  = (const float*)d_in[2];
    const float* b_ih  = (const float*)d_in[3];
    const float* b_hh  = (const float*)d_in[4];
    const float* W_out = (const float*)d_in[5];
    const float* b_out = (const float*)d_in[6];
    float* out = (float*)d_out;

    dim3 grid(B_TOTAL / 16), block(32);
    hipLaunchKernelGGL(gru_wmma_kernel, grid, block, 0, stream,
                       xin, W_ih, W_hh, b_ih, b_hh, W_out, b_out, out);
}
